// HeteroMoELayer_67577015435319
// MI455X (gfx1250) — matrix operand order repaired and checked by reading, not compiled, not run on hardware
//
#include <hip/hip_runtime.h>
#include <cstdint>
#include <type_traits>

typedef _Float16 half16 __attribute__((ext_vector_type(16)));
typedef _Float16 half8  __attribute__((ext_vector_type(8)));
typedef float    float8 __attribute__((ext_vector_type(8)));

#define C_CH 256
#define HW   4096
#define FSTR 264   // padded halfs per pixel row in LDS feature tiles (bank-conflict free)

__device__ __forceinline__ float gelu_exact(float z) {
    return 0.5f * z * (1.0f + erff(z * 0.70710678118654752f));
}

// B-fragment: lane L wants f16 channels kb*32 + (L<16?0:16) + [0..15] at pixel p=L%16,
// which is 32 contiguous bytes in the [pixel][channel] LDS tile (16B-aligned).
__device__ __forceinline__ half16 ld_bfrag(const _Float16* ptr) {
    half8 lo = *(const half8*)ptr;
    half8 hi = *(const half8*)(ptr + 8);
    half16 r;
#pragma unroll
    for (int i = 0; i < 8; ++i) { r[i] = lo[i]; r[i + 8] = hi[i]; }
    return r;
}

// ---------------- pooling: one block per (b,c) plane ----------------
__global__ __launch_bounds__(256) void k_pool(const float* __restrict__ x,
                                              float* __restrict__ pooled) {
    const int bc = blockIdx.x;                       // 0..8191
    const float4* p4 = (const float4*)(x + (size_t)bc * HW);
    const int t = threadIdx.x;
    float4 a = p4[t], b = p4[t + 256], c = p4[t + 512], d = p4[t + 768];
    float s = a.x + a.y + a.z + a.w + b.x + b.y + b.z + b.w +
              c.x + c.y + c.z + c.w + d.x + d.y + d.z + d.w;
    __shared__ float red[256];
    red[t] = s;
    __syncthreads();
    for (int off = 128; off > 0; off >>= 1) {
        if (t < off) red[t] += red[t + off];
        __syncthreads();
    }
    if (t == 0) pooled[bc] = red[0] * (1.0f / 4096.0f);
}

// ---------------- routing: argmax of 3 logits per batch ----------------
__global__ void k_route(const float* __restrict__ pooled, const float* __restrict__ rw,
                        const float* __restrict__ rb, int* __restrict__ idx) {
    const int b = threadIdx.x;
    if (b >= 32) return;
    const float* pp = pooled + b * 256;
    float l0 = rb[0], l1 = rb[1], l2 = rb[2];
    for (int c = 0; c < 256; ++c) {
        const float pv = pp[c];
        l0 += rw[c] * pv;
        l1 += rw[256 + c] * pv;
        l2 += rw[512 + c] * pv;
    }
    int best = 0; float bv = l0;
    if (l1 > bv) { bv = l1; best = 1; }   // strict > : ties keep lowest index (top_k)
    if (l2 > bv) { bv = l2; best = 2; }
    idx[b] = best;
}

// ---------------- prep: f16 A-fragment swizzle of weights + folded BN ----------------
__global__ __launch_bounds__(256) void k_prep(
    const float* __restrict__ pw0, const float* __restrict__ pw1,
    const float* __restrict__ pw2, const float* __restrict__ e2k,
    const float* __restrict__ g0, const float* __restrict__ bb0,
    const float* __restrict__ m0, const float* __restrict__ v0,
    const float* __restrict__ g1, const float* __restrict__ bb1,
    const float* __restrict__ m1, const float* __restrict__ v1,
    const float* __restrict__ g2, const float* __restrict__ bb2,
    const float* __restrict__ m2, const float* __restrict__ v2,
    _Float16* __restrict__ wfr, float* __restrict__ bn) {
    const int t = blockIdx.x * 256 + threadIdx.x;
    if (t < 16384) {                                  // 4 matrices × 16 mt × 8 kb × 32 lanes
        const int lane = t & 31;
        const int kb   = (t >> 5) & 7;
        const int mt   = (t >> 8) & 15;
        const int m    = t >> 12;
        const float* W = (m == 0) ? pw0 : (m == 1) ? pw1 : (m == 2) ? pw2 : e2k;
        const int o    = mt * 16 + (lane & 15);       // A: M = lane%16
        const int koff = (lane < 16) ? 0 : 16;
        _Float16* dst  = wfr + m * 65536 + (mt * 8 + kb) * 512 + lane * 16;
#pragma unroll
        for (int e = 0; e < 16; ++e) {                // A: K(e) = 16*(lane/16) + e, contiguous (matches B)
            const int K = e + koff;
            dst[e] = (_Float16)W[o * 256 + kb * 32 + K];
        }
    } else {
        const int u = t - 16384;
        if (u < 768) {                                // folded BN: scale = g*rsqrt(v+eps)
            const int e = u >> 8;
            const int c = u & 255;
            const float* g  = (e == 0) ? g0  : (e == 1) ? g1  : g2;
            const float* be = (e == 0) ? bb0 : (e == 1) ? bb1 : bb2;
            const float* me = (e == 0) ? m0  : (e == 1) ? m1  : m2;
            const float* ve = (e == 0) ? v0  : (e == 1) ? v1  : v2;
            const float sc = g[c] * rsqrtf(ve[c] + 1e-5f);
            bn[e * 512 + c]       = sc;
            bn[e * 512 + 256 + c] = be[c] - me[c] * sc;
        }
    }
}

// ---------------- fused selected-expert kernel ----------------
__global__ __launch_bounds__(256) void k_main(
    const float* __restrict__ x,
    const float* __restrict__ e0k, const float* __restrict__ e1k,
    const float* __restrict__ pb0, const float* __restrict__ pb1, const float* __restrict__ pb2,
    const _Float16* __restrict__ wfr, const float* __restrict__ bn,
    const int* __restrict__ idx, float* __restrict__ out) {
    __shared__ _Float16 ldsX[16 * FSTR];          //  8448 B : one 16-px feature tile
    __shared__ _Float16 ldsF[4 * 16 * FSTR];      // 33792 B : 4-tile GEMM1 output (expert 2)
    __shared__ float sScale[C_CH], sShift[C_CH], sBias[C_CH];

    const int b    = blockIdx.y;
    const int h0   = blockIdx.x * 4;              // 4 output rows per block
    const int tid  = threadIdx.x;
    const int lane = tid & 31;
    const int wave = tid >> 5;                    // 8 waves
    const int expert = idx[b];                    // block-uniform

    {
        const float* pb = (expert == 0) ? pb0 : (expert == 1) ? pb1 : pb2;
        sScale[tid] = bn[expert * 512 + tid];
        sShift[tid] = bn[expert * 512 + 256 + tid];
        sBias[tid]  = pb[tid];
    }
    __syncthreads();

    const int p    = lane & 15;                   // B-frag / D pixel column
    const int hsel = (lane < 16) ? 0 : 16;        // B-frag K offset
    const int msel = (lane < 16) ? 0 : 8;         // C/D M offset
    const size_t xbase = (size_t)b * C_CH * HW;

    half16 A[2][8];                               // per-wave A fragments: 2 M-tiles × K=256

    if (expert == 2) {
        const _Float16* W1 = wfr + 3 * 65536;     // e2_k   fragments
        const _Float16* W2 = wfr + 2 * 65536;     // e2_pw  fragments
        for (int chunk = 0; chunk < 4; ++chunk) {
            // ---- phase 1: ldsF = gelu(bn(e2_k @ x)) over 4 pixel tiles ----
#pragma unroll
            for (int mtl = 0; mtl < 2; ++mtl)
#pragma unroll
                for (int kb = 0; kb < 8; ++kb)
                    A[mtl][kb] = *(const half16*)(W1 + (((wave * 2 + mtl) * 8 + kb) * 512 + lane * 16));
            for (int tt = 0; tt < 4; ++tt) {
                const int t  = chunk * 4 + tt;
                const int h  = h0 + (t >> 2);
                const int w0 = (t & 3) * 16;
                {   // stage x tile (16 px × 256 ch) into ldsX as [pixel][channel] f16
                    const float* xr = x + xbase + (size_t)tid * HW + (h * 64 + w0);
                    float4 r0 = *(const float4*)(xr);
                    float4 r1 = *(const float4*)(xr + 4);
                    float4 r2 = *(const float4*)(xr + 8);
                    float4 r3 = *(const float4*)(xr + 12);
                    float v[16] = {r0.x, r0.y, r0.z, r0.w, r1.x, r1.y, r1.z, r1.w,
                                   r2.x, r2.y, r2.z, r2.w, r3.x, r3.y, r3.z, r3.w};
#pragma unroll
                    for (int pp = 0; pp < 16; ++pp)
                        ldsX[pp * FSTR + tid] = (_Float16)v[pp];
                }
                __syncthreads();
#pragma unroll
                for (int mtl = 0; mtl < 2; ++mtl) {
                    const int mt = wave * 2 + mtl;
                    float8 acc = {};
#pragma unroll
                    for (int kb = 0; kb < 8; ++kb) {
                        half16 bf = ld_bfrag(&ldsX[p * FSTR + kb * 32 + hsel]);
                        acc = __builtin_amdgcn_wmma_f32_16x16x32_f16(
                                  false, A[mtl][kb], false, bf, (short)0, acc, false, false);
                    }
                    half8 hv;
#pragma unroll
                    for (int v = 0; v < 8; ++v) {
                        const int o = mt * 16 + msel + v;
                        hv[v] = (_Float16)gelu_exact(acc[v] * sScale[o] + sShift[o]);
                    }
                    *(half8*)&ldsF[(tt * 16 + p) * FSTR + mt * 16 + msel] = hv;
                }
                __syncthreads();
            }
            // ---- phase 2: out = e2_pw @ ldsF + pb ----
#pragma unroll
            for (int mtl = 0; mtl < 2; ++mtl)
#pragma unroll
                for (int kb = 0; kb < 8; ++kb)
                    A[mtl][kb] = *(const half16*)(W2 + (((wave * 2 + mtl) * 8 + kb) * 512 + lane * 16));
            for (int tt = 0; tt < 4; ++tt) {
                const int t  = chunk * 4 + tt;
                const int h  = h0 + (t >> 2);
                const int w0 = (t & 3) * 16;
                const size_t obase = xbase + (size_t)(h * 64 + w0 + p);
#pragma unroll
                for (int mtl = 0; mtl < 2; ++mtl) {
                    const int mt = wave * 2 + mtl;
                    float8 acc = {};
#pragma unroll
                    for (int kb = 0; kb < 8; ++kb) {
                        half16 bf = ld_bfrag(&ldsF[(tt * 16 + p) * FSTR + kb * 32 + hsel]);
                        acc = __builtin_amdgcn_wmma_f32_16x16x32_f16(
                                  false, A[mtl][kb], false, bf, (short)0, acc, false, false);
                    }
#pragma unroll
                    for (int v = 0; v < 8; ++v) {
                        const int o = mt * 16 + msel + v;
                        out[obase + (size_t)o * HW] = acc[v] + sBias[o];
                    }
                }
            }
            __syncthreads();   // before next chunk overwrites ldsX/ldsF
        }
    } else {
        // experts 0/1: depthwise 3x3 (dil = expert+1) -> BN -> GELU -> pointwise GEMM
        auto run_dw = [&](auto dilc) {
            constexpr int DIL = decltype(dilc)::value;
            const float* dk = (DIL == 1) ? e0k : e1k;
            const _Float16* Wp = wfr + (DIL - 1) * 65536;
#pragma unroll
            for (int mtl = 0; mtl < 2; ++mtl)
#pragma unroll
                for (int kb = 0; kb < 8; ++kb)
                    A[mtl][kb] = *(const half16*)(Wp + (((wave * 2 + mtl) * 8 + kb) * 512 + lane * 16));
            float kk[9];
#pragma unroll
            for (int j = 0; j < 9; ++j) kk[j] = dk[tid * 9 + j];
            const float* xc = x + xbase + (size_t)tid * HW;   // this thread's channel plane
            for (int t = 0; t < 16; ++t) {
                const int h  = h0 + (t >> 2);
                const int w0 = (t & 3) * 16;
                float accp[16];
#pragma unroll
                for (int pp = 0; pp < 16; ++pp) accp[pp] = 0.f;
#pragma unroll
                for (int ky = 0; ky < 3; ++ky) {
                    const int iy = h + (ky - 1) * DIL;
                    if (iy >= 0 && iy < 64) {
                        const float* xr = xc + iy * 64;
                        float seg[16 + 2 * DIL];
#pragma unroll
                        for (int j = 0; j < 16 + 2 * DIL; ++j) {
                            const int ix = w0 - DIL + j;
                            seg[j] = (ix >= 0 && ix < 64) ? xr[ix] : 0.f;
                        }
#pragma unroll
                        for (int kx = 0; kx < 3; ++kx) {
                            const float kw = kk[ky * 3 + kx];
#pragma unroll
                            for (int pp = 0; pp < 16; ++pp)
                                accp[pp] += kw * seg[pp + kx * DIL];
                        }
                    }
                }
                const float sc = sScale[tid], sh = sShift[tid];
#pragma unroll
                for (int pp = 0; pp < 16; ++pp)
                    ldsX[pp * FSTR + tid] = (_Float16)gelu_exact(accp[pp] * sc + sh);
                __syncthreads();
                const size_t obase = xbase + (size_t)(h * 64 + w0 + p);
#pragma unroll
                for (int mtl = 0; mtl < 2; ++mtl) {
                    const int mt = wave * 2 + mtl;
                    float8 acc = {};
#pragma unroll
                    for (int kb = 0; kb < 8; ++kb) {
                        half16 bf = ld_bfrag(&ldsX[p * FSTR + kb * 32 + hsel]);
                        acc = __builtin_amdgcn_wmma_f32_16x16x32_f16(
                                  false, A[mtl][kb], false, bf, (short)0, acc, false, false);
                    }
#pragma unroll
                    for (int v = 0; v < 8; ++v) {
                        const int o = mt * 16 + msel + v;
                        out[obase + (size_t)o * HW] = acc[v] + sBias[o];
                    }
                }
                __syncthreads();
            }
        };
        if (expert == 0) run_dw(std::integral_constant<int, 1>{});
        else             run_dw(std::integral_constant<int, 2>{});
    }
}

extern "C" void kernel_launch(void* const* d_in, const int* in_sizes, int n_in,
                              void* d_out, int out_size, void* d_ws, size_t ws_size,
                              hipStream_t stream) {
    (void)in_sizes; (void)n_in; (void)out_size; (void)ws_size;
    const float* x   = (const float*)d_in[0];
    const float* rw  = (const float*)d_in[1];
    const float* rb  = (const float*)d_in[2];
    const float* e0k = (const float*)d_in[3];
    const float* g0  = (const float*)d_in[4];
    const float* b0  = (const float*)d_in[5];
    const float* m0  = (const float*)d_in[6];
    const float* v0  = (const float*)d_in[7];
    const float* pw0 = (const float*)d_in[8];
    const float* pb0 = (const float*)d_in[9];
    const float* e1k = (const float*)d_in[10];
    const float* g1  = (const float*)d_in[11];
    const float* b1  = (const float*)d_in[12];
    const float* m1  = (const float*)d_in[13];
    const float* v1  = (const float*)d_in[14];
    const float* pw1 = (const float*)d_in[15];
    const float* pb1 = (const float*)d_in[16];
    const float* e2k = (const float*)d_in[17];
    const float* g2  = (const float*)d_in[18];
    const float* b2  = (const float*)d_in[19];
    const float* m2  = (const float*)d_in[20];
    const float* v2  = (const float*)d_in[21];
    const float* pw2 = (const float*)d_in[22];
    const float* pb2 = (const float*)d_in[23];

    char* wsB       = (char*)d_ws;
    float* pooled   = (float*)wsB;                 // 32*256 f32      = 32768 B
    int*   idx      = (int*)(wsB + 32768);         // 32 i32          =   128 B
    float* bnf      = (float*)(wsB + 33024);       // 3*512 f32       =  6144 B
    _Float16* wfr   = (_Float16*)(wsB + 40960);    // 4*65536 f16     = 524288 B

    k_pool <<<32 * 256, 256, 0, stream>>>(x, pooled);
    k_route<<<1, 32, 0, stream>>>(pooled, rw, rb, idx);
    k_prep <<<67, 256, 0, stream>>>(pw0, pw1, pw2, e2k,
                                    g0, b0, m0, v0,
                                    g1, b1, m1, v1,
                                    g2, b2, m2, v2,
                                    wfr, bnf);
    dim3 grid(16, 32);   // (h-block, batch)
    k_main <<<grid, 256, 0, stream>>>(x, e0k, e1k, pb0, pb1, pb2, wfr, bnf, idx, (float*)d_out);
}
